// CONV_3x3rand_71657234366728
// MI455X (gfx1250) — compile-verified, matrix operand order checked by reading, not run on hardware
//
#include <hip/hip_runtime.h>

typedef _Float16 v16h __attribute__((ext_vector_type(16)));
typedef _Float16 h8   __attribute__((ext_vector_type(8)));
typedef float    v8f  __attribute__((ext_vector_type(8)));

#define HWSZ   3136            // 56*56
#define WDIM   56
#define CIN    128
#define COUT   256
#define NBATCH 32
#define KTOT   1152            // 9 * 128
#define NPOS   (NBATCH * HWSZ) // 100352
#define ASTR   40              // padded LDS row stride (f16), 80B rows, 16B aligned
#define NKC    36              // K chunks of 32

// ---- CDNA5 async global->LDS (ASYNCcnt-tracked), 16B per lane ----
__device__ __forceinline__ void async_b128(unsigned lds_off, const void* g) {
  asm volatile("global_load_async_to_lds_b128 %0, %1, off"
               :: "v"(lds_off), "v"((unsigned long long)(uintptr_t)g)
               : "memory");
}
__device__ __forceinline__ void wait_async0() {
  asm volatile("s_wait_asynccnt 0" ::: "memory");
}

// ---------------- prep: permute + NCHW->NHWC + f32->f16 ----------------
__global__ __launch_bounds__(256) void prep_x_kernel(
    const float* __restrict__ x, const int* __restrict__ perm,
    _Float16* __restrict__ xp) {
  const int t    = threadIdx.x;
  const int pair = t >> 7;
  const int c    = t & 127;
  const int ns   = blockIdx.x * 2 + pair;   // 0..100351
  const int n    = ns / HWSZ;
  const int s    = ns - n * HWSZ;
  const int ps   = perm[s];
  const float v  = x[(size_t)(n * CIN + c) * HWSZ + ps];
  xp[(size_t)ns * CIN + c] = (_Float16)v;
}

// ---------------- prep: weight -> f16 [co][tap*128 + ci] ----------------
__global__ __launch_bounds__(256) void prep_w_kernel(
    const float* __restrict__ w, _Float16* __restrict__ wt) {
  const int e   = blockIdx.x * 256 + threadIdx.x;   // < 294912
  const int co  = e / KTOT;
  const int k   = e - co * KTOT;
  const int tap = k >> 7;
  const int ci  = k & 127;
  const int kh  = tap / 3;
  const int kw  = tap - kh * 3;
  wt[e] = (_Float16)w[(((size_t)(co * CIN + ci) * 3 + kh) * 3 + kw)];
}

// ---------------- implicit-GEMM conv via WMMA, async double-buffered ----------------
// grid(784, 2); block tile: 128 positions x 128 channels; 8 waves of 32x64.
__global__ __launch_bounds__(256) void conv_wmma_kernel(
    const _Float16* __restrict__ xp, const _Float16* __restrict__ wt,
    float* __restrict__ out) {
  __shared__ __align__(16) char smem[40960];
  // double-buffered K stage:
  //   A[buf]: 128 x ASTR f16 @ smem + buf*10240         (positions x 32ci)
  //   B[buf]: 128 x ASTR f16 @ smem + 20480 + buf*10240 (channels  x 32k)
  // output transpose reuses smem[0..33791] as 64 x 132 f32.

  const int bm  = blockIdx.x;   // position block (0..783)
  const int cb  = blockIdx.y;   // channel block  (0..1)
  const int tid = threadIdx.x;

  // --- A-tile loader coords (loop-invariant geometry) ---
  const int p_row = tid & 127;
  const int segA  = tid >> 7;                 // ci offset 16*segA
  const int pos   = bm * 128 + p_row;
  const int n     = pos / HWSZ;
  const int hw    = pos - n * HWSZ;
  const int oh    = hw / WDIM;
  const int ow    = hw - oh * WDIM;

  // --- B-tile loader coords ---
  const int b_row = tid & 127;                // local channel 0..127
  const int segB  = tid >> 7;                 // k offset 16*segB

  // --- wave / fragment coords ---
  const int wv   = tid >> 5;
  const int lane = tid & 31;
  const int wm   = wv & 3;                    // pos sub-block (0..3) -> rows wm*32
  const int wn   = wv >> 2;                   // ch  sub-block (0..1) -> cols wn*64
  const int ll   = lane & 15;
  const int kh2  = lane >> 4;

  // stage chunk kc into buffer buf (async global->LDS; zeros for OOB taps)
  auto stage = [&](int kc, int buf) {
    _Float16* Ab = (_Float16*)(smem + buf * 10240);
    _Float16* Bb = (_Float16*)(smem + 20480 + buf * 10240);
    const int tap = kc >> 2;
    const int cio = (kc & 3) * 32 + segA * 16;
    const int kh  = tap / 3;
    const int kw  = tap - kh * 3;
    const int ih  = oh + kh - 1;
    const int iw  = ow + kw - 1;
    _Float16* adst = Ab + p_row * ASTR + segA * 16;
    const unsigned aoff = (unsigned)(uintptr_t)adst;
    if ((unsigned)ih < 56u && (unsigned)iw < 56u) {
      const _Float16* src = xp + ((size_t)(n * HWSZ + ih * WDIM + iw) * CIN + cio);
      async_b128(aoff,      src);
      async_b128(aoff + 16, src + 8);
    } else {
      const uint4 z = make_uint4(0u, 0u, 0u, 0u);
      ((uint4*)adst)[0] = z;
      ((uint4*)adst)[1] = z;
    }
    _Float16* bdst = Bb + b_row * ASTR + segB * 16;
    const unsigned boff = (unsigned)(uintptr_t)bdst;
    const _Float16* bsrc =
        wt + ((size_t)(cb * 128 + b_row) * KTOT + kc * 32 + segB * 16);
    async_b128(boff,      bsrc);
    async_b128(boff + 16, bsrc + 8);
  };

  v8f acc[2][4];
#pragma unroll
  for (int i = 0; i < 2; ++i)
#pragma unroll
    for (int j = 0; j < 4; ++j)
      acc[i][j] = (v8f){0.f, 0.f, 0.f, 0.f, 0.f, 0.f, 0.f, 0.f};

  stage(0, 0);   // prologue

  for (int kc = 0; kc < NKC; ++kc) {
    const int buf = kc & 1;
    wait_async0();        // my async writes into buf are done
    __syncthreads();      // everyone's are; buf^1 readers from kc-1 also done
    if (kc + 1 < NKC) stage(kc + 1, buf ^ 1);

    const _Float16* Ab = (const _Float16*)(smem + buf * 10240);
    const _Float16* Bb = (const _Float16*)(smem + 20480 + buf * 10240);

    v16h a[2], b[4];
#pragma unroll
    for (int i = 0; i < 2; ++i) {
      const _Float16* base = Ab + (wm * 32 + i * 16 + ll) * ASTR + kh2 * 8;
      h8 lo = *(const h8*)base;
      h8 hi = *(const h8*)(base + 16);
      a[i] = __builtin_shufflevector(lo, hi, 0, 1, 2, 3, 4, 5, 6, 7,
                                     8, 9, 10, 11, 12, 13, 14, 15);
    }
#pragma unroll
    for (int j = 0; j < 4; ++j) {
      const _Float16* base = Bb + (wn * 64 + j * 16 + ll) * ASTR + kh2 * 8;
      h8 lo = *(const h8*)base;
      h8 hi = *(const h8*)(base + 16);
      b[j] = __builtin_shufflevector(lo, hi, 0, 1, 2, 3, 4, 5, 6, 7,
                                     8, 9, 10, 11, 12, 13, 14, 15);
    }
#pragma unroll
    for (int i = 0; i < 2; ++i)
#pragma unroll
      for (int j = 0; j < 4; ++j)
        acc[i][j] = __builtin_amdgcn_wmma_f32_16x16x32_f16(
            false, a[i], false, b[j], (short)0, acc[i][j], false, false);
  }

  // ---- transpose + coalesced NCHW store, two 64-channel passes ----
  float* Cs = (float*)smem;   // 64 x 132 f32
#pragma unroll 1
  for (int half = 0; half < 2; ++half) {
    __syncthreads();
    if (wn == half) {
#pragma unroll
      for (int i = 0; i < 2; ++i)
#pragma unroll
        for (int j = 0; j < 4; ++j) {
          const int ch_l    = j * 16 + ll;                 // 0..63 within half
          const int pr_base = wm * 32 + i * 16 + kh2 * 8;
#pragma unroll
          for (int r = 0; r < 8; ++r)
            Cs[ch_l * 132 + pr_base + r] = acc[i][j][r];
        }
    }
    __syncthreads();
#pragma unroll 4
    for (int it = 0; it < 32; ++it) {
      const int idx  = it * 256 + tid;
      const int ch_l = idx >> 7;
      const int p_l  = idx & 127;
      const int s    = bm * 128 + p_l;
      const int nn   = s / HWSZ;
      const int shw  = s - nn * HWSZ;
      out[(size_t)(nn * COUT + cb * 128 + half * 64 + ch_l) * HWSZ + shw] =
          Cs[ch_l * 132 + p_l];
    }
  }
}

// ---------------- BN stats: one block per channel ----------------
__global__ __launch_bounds__(256) void bn_stats_kernel(
    const float* __restrict__ out, float* __restrict__ stats) {
  const int c   = blockIdx.x;   // 0..255
  const int tid = threadIdx.x;
  float s = 0.f, s2 = 0.f;
  for (int idx = tid; idx < NPOS; idx += 256) {
    const int n  = idx / HWSZ;
    const int hw = idx - n * HWSZ;
    const float v = out[(size_t)(n * COUT + c) * HWSZ + hw];
    s  += v;
    s2 += v * v;
  }
  __shared__ float rs[256], rs2[256];
  rs[tid]  = s;
  rs2[tid] = s2;
  __syncthreads();
  for (int off = 128; off > 0; off >>= 1) {
    if (tid < off) {
      rs[tid]  += rs[tid + off];
      rs2[tid] += rs2[tid + off];
    }
    __syncthreads();
  }
  if (tid == 0) {
    stats[c]       = rs[0];
    stats[256 + c] = rs2[0];
  }
}

// ---------------- BN apply + ReLU, in place, float4 ----------------
__global__ __launch_bounds__(256) void bn_apply_kernel(
    float* __restrict__ out, const float* __restrict__ stats,
    const float* __restrict__ gamma, const float* __restrict__ beta) {
  const size_t e = (size_t)blockIdx.x * 1024 + (size_t)threadIdx.x * 4;
  const int c    = (int)((e / HWSZ) % COUT);
  const float inv_n = 1.0f / (float)NPOS;
  const float mean  = stats[c] * inv_n;
  const float var   = stats[256 + c] * inv_n - mean * mean;
  const float scale = rsqrtf(var + 1e-5f) * gamma[c];
  const float shift = beta[c] - mean * scale;
  float4 v = *(float4*)(out + e);
  v.x = fmaxf(v.x * scale + shift, 0.f);
  v.y = fmaxf(v.y * scale + shift, 0.f);
  v.z = fmaxf(v.z * scale + shift, 0.f);
  v.w = fmaxf(v.w * scale + shift, 0.f);
  *(float4*)(out + e) = v;
}

extern "C" void kernel_launch(void* const* d_in, const int* in_sizes, int n_in,
                              void* d_out, int out_size, void* d_ws, size_t ws_size,
                              hipStream_t stream) {
  const float* x     = (const float*)d_in[0];
  const float* w     = (const float*)d_in[1];
  const float* gamma = (const float*)d_in[2];
  const float* beta  = (const float*)d_in[3];
  const int*   perm  = (const int*)d_in[4];
  float* out = (float*)d_out;

  char* ws = (char*)d_ws;
  _Float16* xp    = (_Float16*)ws;                          // 25,690,112 B
  _Float16* wt    = (_Float16*)(ws + 25690112);             //    589,824 B
  float*    stats = (float*)(ws + 25690112 + 589824);       //      2,048 B

  prep_x_kernel<<<NPOS / 2, 256, 0, stream>>>(x, perm, xp);
  prep_w_kernel<<<(COUT * KTOT) / 256, 256, 0, stream>>>(w, wt);
  conv_wmma_kernel<<<dim3(NPOS / 128, COUT / 128), 256, 0, stream>>>(xp, wt, out);
  bn_stats_kernel<<<COUT, 256, 0, stream>>>(out, stats);
  bn_apply_kernel<<<(NBATCH * COUT * HWSZ) / 1024, 256, 0, stream>>>(out, stats, gamma, beta);
}
